// DiscretizedManifoldBlock_24068996727161
// MI455X (gfx1250) — matrix operand order repaired and checked by reading, not compiled, not run on hardware
//
#include <hip/hip_runtime.h>
#include <hip/hip_bf16.h>
#include <math.h>

typedef __bf16 bf16;
typedef __attribute__((ext_vector_type(16))) __bf16 v16bf;
typedef __attribute__((ext_vector_type(8)))  __bf16 v8bf;
typedef __attribute__((ext_vector_type(8)))  float  v8f;

constexpr int BB = 4, T = 4096, C = 1024, H = 16, CS = 256, D = 64, NC = 16;
constexpr int LVL = 4, KC = 1024;
constexpr int NT = BB * T;                 // 16384 tokens
constexpr size_t NTC = (size_t)NT * C;     // 16,777,216 elems
constexpr float EPS = 1e-5f;

// ---------------------------------------------------------------------------
// Generic bf16 WMMA GEMM:  out = act(alpha * (A @ Wt^T) + bias)
// A: [M,K] bf16 row-major.  Wt: [N,K] bf16 row-major (pre-transposed weight).
// Each wave computes a 16x64 strip: 4 accumulators share one A fragment per
// K-step (2.5 b128 loads per WMMA instead of 4).  8 waves/block -> 128x64.
// Fragment packing (same contiguous k order for A and B):
//   A lane l (row m0+(l&15)): elems 0-15 = K[h*16 .. h*16+15],
//   B lane l (col n +(l&15)): elems 0-15 = K[h*16 .. h*16+15],  h = l>>4.
// act: 0=none 1=sigmoid 2=gelu(exact).  store_mode 1 = ho2 transposed scatter.
// ---------------------------------------------------------------------------
__global__ __launch_bounds__(256) void k_gemm_bf16(
    const bf16* __restrict__ A, const bf16* __restrict__ Wt,
    const float* __restrict__ bias, float alpha,
    float* __restrict__ Cf, bf16* __restrict__ Cb,
    int Mm, int Nn, int Kk, int act, int store_mode)
{
  const int wave = threadIdx.x >> 5;
  const int lane = threadIdx.x & 31;
  const int m0 = (blockIdx.y * 8 + wave) * 16;
  const int n0 = blockIdx.x * 64;
  if (m0 >= Mm) return;
  const int hsel = lane >> 4;   // 0 or 1
  const int l16  = lane & 15;
  const bf16* aptr = A + (size_t)(m0 + l16) * Kk + hsel * 16;
  const bf16* bptr[4];
#pragma unroll
  for (int j = 0; j < 4; ++j)
    bptr[j] = Wt + (size_t)(n0 + j * 16 + l16) * Kk + hsel * 16;

  v8f acc[4] = {};
  for (int k0 = 0; k0 < Kk; k0 += 32) {
    __builtin_prefetch(aptr + k0 + 128);           // global_prefetch hint
    v8bf a_lo = *(const v8bf*)(aptr + k0);
    v8bf a_hi = *(const v8bf*)(aptr + k0 + 8);
    v16bf a;
#pragma unroll
    for (int i = 0; i < 8; ++i) { a[i] = a_lo[i]; a[i + 8] = a_hi[i]; }
#pragma unroll
    for (int j = 0; j < 4; ++j) {
      v16bf b = *(const v16bf*)(bptr[j] + k0);
      acc[j] = __builtin_amdgcn_wmma_f32_16x16x32_bf16(
          false, a, false, b, (short)0, acc[j], false, false);
    }
  }

#pragma unroll
  for (int j = 0; j < 4; ++j) {
    const int col = n0 + j * 16 + l16;
    const float bv = bias ? bias[col] : 0.f;
#pragma unroll
    for (int i = 0; i < 8; ++i) {
      int row = m0 + hsel * 8 + i;                 // C/D layout: vgpr i, half
      float v = alpha * acc[j][i] + bv;
      if (act == 1)      v = 1.f / (1.f + __expf(-v));
      else if (act == 2) v = 0.5f * v * (1.f + erff(v * 0.70710678118654752f));
      if (store_mode == 0) {
        size_t o = (size_t)row * Nn + col;
        if (Cf) Cf[o] = v;
        if (Cb) Cb[o] = (bf16)v;
      } else {                                     // ho2: (b,h,nc,cs,d)->(b,t,c)
        int cs = row & (CS - 1); int t1 = row >> 8;
        int nc = t1 & (NC - 1);  int t2 = t1 >> 4;
        int hh = t2 & (H - 1);   int bq = t2 >> 4;
        size_t o = ((size_t)(bq * T + nc * CS + cs)) * C + hh * D + col;
        Cb[o] = (bf16)v;
      }
    }
  }
}

// LayerNorm over C=1024; optional residuals; optional f32 / bf16 outputs.
__global__ __launch_bounds__(256) void k_ln1024(
    const float* __restrict__ X, const float* __restrict__ g,
    const float* __restrict__ bpar,
    const float* __restrict__ res1, const float* __restrict__ res2,
    float* __restrict__ outF, bf16* __restrict__ outB)
{
  const int row = blockIdx.x, tid = threadIdx.x;
  const size_t base = (size_t)row * C;
  float v[4]; float s = 0.f;
#pragma unroll
  for (int i = 0; i < 4; ++i) { v[i] = X[base + tid + i * 256]; s += v[i]; }
  __shared__ float red[256];
  red[tid] = s; __syncthreads();
  for (int off = 128; off > 0; off >>= 1) { if (tid < off) red[tid] += red[tid + off]; __syncthreads(); }
  const float mean = red[0] * (1.f / C);
  __syncthreads();
  float var = 0.f;
#pragma unroll
  for (int i = 0; i < 4; ++i) { float d = v[i] - mean; var += d * d; }
  red[tid] = var; __syncthreads();
  for (int off = 128; off > 0; off >>= 1) { if (tid < off) red[tid] += red[tid + off]; __syncthreads(); }
  const float rstd = rsqrtf(red[0] * (1.f / C) + EPS);
#pragma unroll
  for (int i = 0; i < 4; ++i) {
    int colx = tid + i * 256;
    float y = (v[i] - mean) * rstd * g[colx] + bpar[colx];
    if (res1) y += res1[base + colx];
    if (res2) y += res2[base + colx];
    if (outF) outF[base + colx] = y;
    if (outB) outB[base + colx] = (bf16)y;
  }
}

// wave-level LN over D=64 (2 elems/lane)
__device__ inline void ln64(float x0, float x1, const float* g, const float* bb,
                            int lane, float& o0, float& o1) {
  float s = x0 + x1;
#pragma unroll
  for (int off = 16; off; off >>= 1) s += __shfl_xor(s, off, 32);
  float mean = s * (1.f / 64.f);
  float d0 = x0 - mean, d1 = x1 - mean;
  float var = d0 * d0 + d1 * d1;
#pragma unroll
  for (int off = 16; off; off >>= 1) var += __shfl_xor(var, off, 32);
  float rstd = rsqrtf(var * (1.f / 64.f) + EPS);
  o0 = d0 * rstd * g[lane] + bb[lane];
  o1 = d1 * rstd * g[lane + 32] + bb[lane + 32];
}

// elementwise: a *= b
__global__ __launch_bounds__(256) void k_mul(float* a, const float* b, long n) {
  long i = (long)blockIdx.x * 256 + threadIdx.x;
  if (i < n) a[i] *= b[i];
}
// elementwise: o = a + b
__global__ __launch_bounds__(256) void k_add(const float* a, const float* b, float* o, long n) {
  long i = (long)blockIdx.x * 256 + threadIdx.x;
  if (i < n) o[i] = a[i] + b[i];
}

// in-place inclusive cumsum over cs within each (b,h,nc,d) series + chunk sums
__global__ __launch_bounds__(256) void k_cumsum(float* gm, float* csum) {
  int s = blockIdx.x * 256 + threadIdx.x;        // < B*H*NC*D = 65536
  int d = s & 63, nc = (s >> 6) & 15, h = (s >> 10) & 15, b = s >> 14;
  size_t base = ((size_t)(b * T + nc * CS)) * C + h * D + d;
  float acc = 0.f;
  for (int cs = 0; cs < CS; ++cs) { size_t a = base + (size_t)cs * C; acc += gm[a]; gm[a] = acc; }
  csum[s] = acc;
}

// exclusive cumsum of chunk sums over nc
__global__ __launch_bounds__(256) void k_carry(const float* csum, float* carr) {
  int t = blockIdx.x * 256 + threadIdx.x;        // < B*H*D = 4096
  int d = t & 63, h = (t >> 6) & 15, b = t >> 10;
  float acc = 0.f;
  for (int nc = 0; nc < NC; ++nc) {
    int idx = ((b * H + h) * NC + nc) * D + d;
    carr[idx] = acc; acc += csum[idx];
  }
}

// ncar = LN64(carries)
__global__ __launch_bounds__(256) void k_ncar(const float* carr, const float* g,
                                              const float* bb, float* ncar) {
  int wave = threadIdx.x >> 5, lane = threadIdx.x & 31;
  int row = blockIdx.x * 8 + wave;               // < B*H*NC = 1024
  size_t base = (size_t)row * D;
  float o0, o1;
  ln64(carr[base + lane], carr[base + lane + 32], g, bb, lane, o0, o1);
  ncar[base + lane] = o0; ncar[base + lane + 32] = o1;
}

// combined[r,0:64]=x_heads ; combined[r,64:128]=LN64(ncar + lcm[cs-1])
__global__ __launch_bounds__(256) void k_combined(
    const float* __restrict__ ln1f, const float* __restrict__ lcm,
    const float* __restrict__ ncar, const float* __restrict__ cg,
    const float* __restrict__ cb, bf16* __restrict__ cmb) {
  int wave = threadIdx.x >> 5, lane = threadIdx.x & 31;
  int r = blockIdx.x * 8 + wave;                 // < B*H*NC*CS = 262144
  int cs = r & (CS - 1);
  int nrow = r >> 8;                             // (b*H+h)*NC+nc
  int nc = nrow & (NC - 1); int bh = nrow >> 4;
  int h = bh & (H - 1);     int b = bh >> 4;
  size_t xbase = ((size_t)(b * T + nc * CS + cs)) * C + h * D;
  float x0 = ln1f[xbase + lane], x1 = ln1f[xbase + lane + 32];
  float c0 = ncar[(size_t)nrow * D + lane], c1 = ncar[(size_t)nrow * D + lane + 32];
  if (cs > 0) { c0 += lcm[xbase - C + lane]; c1 += lcm[xbase - C + lane + 32]; }
  float o0, o1; ln64(c0, c1, cg, cb, lane, o0, o1);
  size_t ob = (size_t)r * 128;
  cmb[ob + lane]      = (bf16)x0; cmb[ob + 32 + lane] = (bf16)x1;
  cmb[ob + 64 + lane] = (bf16)o0; cmb[ob + 96 + lane] = (bf16)o1;
}

// wave-per-row argmin over n codes (first-min tie break)
__global__ __launch_bounds__(256) void k_argmin(const float* sc, int* idx, int rows, int n) {
  int wave = threadIdx.x >> 5, lane = threadIdx.x & 31;
  int row = blockIdx.x * 8 + wave; if (row >= rows) return;
  const float* s = sc + (size_t)row * n;
  float best = 3.4e38f; int bi = 0;
  for (int j = lane; j < n; j += 32) { float v = s[j]; if (v < best) { best = v; bi = j; } }
#pragma unroll
  for (int off = 16; off; off >>= 1) {
    float ov = __shfl_xor(best, off, 32); int oi = __shfl_xor(bi, off, 32);
    if (ov < best || (ov == best && oi < bi)) { best = ov; bi = oi; }
  }
  if (lane == 0) idx[row] = bi;
}

// rvq step: q=E[idx]; loss+=|q-res|^2; total+=q; res-=q (f32 + bf16 mirror)
__global__ __launch_bounds__(256) void k_rvq_update(
    const float* __restrict__ E, const int* __restrict__ idx,
    float* __restrict__ res, bf16* __restrict__ resb,
    float* __restrict__ total, float* __restrict__ loss, long n) {
  long i = (long)blockIdx.x * 256 + threadIdx.x;
  float d2 = 0.f;
  if (i < n) {
    int row = (int)(i >> 10), c = (int)(i & 1023);
    float q = E[(size_t)idx[row] * C + c];
    float r = res[i];
    float df = q - r; d2 = df * df;
    total[i] += q;
    float rn = r - q;
    res[i] = rn; resb[i] = (bf16)rn;
  }
  __shared__ float red[256];
  red[threadIdx.x] = d2; __syncthreads();
  for (int off = 128; off; off >>= 1) { if (threadIdx.x < off) red[threadIdx.x] += red[threadIdx.x + off]; __syncthreads(); }
  if (threadIdx.x == 0) atomicAdd(loss, red[0]);
}

// codebook row squared norms
__global__ __launch_bounds__(256) void k_enorm(const float* E, float* en) {
  int row = blockIdx.x;                          // < LVL*KC
  const float* e = E + (size_t)row * C;
  float s = 0.f;
  for (int i = threadIdx.x; i < C; i += 256) { float v = e[i]; s += v * v; }
  __shared__ float red[256];
  red[threadIdx.x] = s; __syncthreads();
  for (int off = 128; off; off >>= 1) { if (threadIdx.x < off) red[threadIdx.x] += red[threadIdx.x + off]; __syncthreads(); }
  if (threadIdx.x == 0) en[row] = red[0];
}

__global__ __launch_bounds__(256) void k_conv_bf16(const float* W, bf16* O, long n) {
  long i = (long)blockIdx.x * 256 + threadIdx.x;
  if (i < n) O[i] = (bf16)W[i];
}
__global__ __launch_bounds__(256) void k_convT_bf16(const float* W, bf16* Wt, int Kk, int Nn) {
  long i = (long)blockIdx.x * 256 + threadIdx.x;
  if (i >= (long)Kk * Nn) return;
  int k = (int)(i / Nn), n = (int)(i % Nn);
  Wt[(size_t)n * Kk + k] = (bf16)W[i];
}
__global__ void k_finalize_loss(const float* loss, float* out, long ntc) {
  if (threadIdx.x == 0) {
    float v = loss[0] / ((float)ntc * (float)LVL);
    out[ntc] = v; out[ntc + 1] = v;
  }
}

// ---------------------------------------------------------------------------
extern "C" void kernel_launch(void* const* d_in, const int* in_sizes, int n_in,
                              void* d_out, int out_size, void* d_ws, size_t ws_size,
                              hipStream_t stream) {
  const float* x     = (const float*)d_in[0];
  const float* ln1g  = (const float*)d_in[1];  const float* ln1b  = (const float*)d_in[2];
  const float* markW = (const float*)d_in[3];  const float* markb = (const float*)d_in[4];
  const float* gateW = (const float*)d_in[5];  const float* gateb = (const float*)d_in[6];
  const float* carg  = (const float*)d_in[7];  const float* carb  = (const float*)d_in[8];
  const float* cdg   = (const float*)d_in[9];  const float* cdb   = (const float*)d_in[10];
  const float* hoW1  = (const float*)d_in[11]; const float* hob1  = (const float*)d_in[12];
  const float* hoW2  = (const float*)d_in[13]; const float* hob2  = (const float*)d_in[14];
  const float* projW = (const float*)d_in[15]; const float* projb = (const float*)d_in[16];
  const float* alng  = (const float*)d_in[17]; const float* alnb  = (const float*)d_in[18];
  const float* ln2g  = (const float*)d_in[19]; const float* ln2b  = (const float*)d_in[20];
  const float* mlpW1 = (const float*)d_in[21]; const float* mlpb1 = (const float*)d_in[22];
  const float* mlpW2 = (const float*)d_in[23]; const float* mlpb2 = (const float*)d_in[24];
  const float* ln3g  = (const float*)d_in[25]; const float* ln3b  = (const float*)d_in[26];
  const float* vq    = (const float*)d_in[27];

  char* ws = (char*)d_ws;
  constexpr size_t MB = 1ull << 20;
  constexpr size_t OFF_BUF1 = 0;          // ln1 f32 (A..I) -> scores (O)
  constexpr size_t OFF_BUF2 = 64 * MB;    // pm/gm/lcm -> proj_out -> mlp2 -> total
  constexpr size_t OFF_BUF3 = 128 * MB;   // gate -> x1
  constexpr size_t OFF_ABF  = 192 * MB;   // ln1 bf16 -> ln2 bf16
  constexpr size_t OFF_HIDR = 224 * MB;   // combined bf16 / mlp hidden bf16 / res f32
  constexpr size_t OFF_H1R  = 352 * MB;   // h1 bf16 -> x2 f32
  constexpr size_t OFF_ATTR = 416 * MB;   // attn bf16 -> res bf16
  constexpr size_t OFF_CSUM = 448 * MB;
  constexpr size_t OFF_CARR = OFF_CSUM + 256 * 1024;
  constexpr size_t OFF_NCAR = OFF_CARR + 256 * 1024;
  constexpr size_t OFF_IDX  = OFF_NCAR + 256 * 1024;
  constexpr size_t OFF_LOSS = OFF_IDX + 64 * 1024;
  constexpr size_t OFF_MARKWT = 449 * MB;
  constexpr size_t OFF_GATEWT = OFF_MARKWT + 2 * MB;
  constexpr size_t OFF_PROJWT = OFF_GATEWT + 2 * MB;
  constexpr size_t OFF_MLP1WT = OFF_PROJWT + 2 * MB;
  constexpr size_t OFF_MLP2WT = OFF_MLP1WT + 8 * MB;
  constexpr size_t OFF_HO1WT  = OFF_MLP2WT + 8 * MB;
  constexpr size_t OFF_HO2WT  = OFF_HO1WT + 64 * 1024;
  constexpr size_t OFF_EBF    = OFF_HO2WT + 64 * 1024;
  constexpr size_t OFF_ENORM  = OFF_EBF + 8 * MB;

  auto F  = [&](size_t o) { return (float*)(ws + o); };
  auto Bf = [&](size_t o) { return (bf16*)(ws + o); };

  auto gemm = [&](const bf16* A, const bf16* Wt, const float* bias, float alpha,
                  float* Cf, bf16* Cb, int M, int N, int K, int act, int mode) {
    dim3 g(N / 64, M / 128);
    k_gemm_bf16<<<g, 256, 0, stream>>>(A, Wt, bias, alpha, Cf, Cb, M, N, K, act, mode);
  };
  auto blocks = [](long n) { return (unsigned)((n + 255) / 256); };

  // ---- weight prep (bf16, transposed to [N,K]) ----
  k_convT_bf16<<<blocks((long)C * C), 256, 0, stream>>>(markW, Bf(OFF_MARKWT), C, C);
  k_convT_bf16<<<blocks((long)C * C), 256, 0, stream>>>(gateW, Bf(OFF_GATEWT), C, C);
  k_convT_bf16<<<blocks((long)C * C), 256, 0, stream>>>(projW, Bf(OFF_PROJWT), C, C);
  k_convT_bf16<<<blocks((long)C * 4 * C), 256, 0, stream>>>(mlpW1, Bf(OFF_MLP1WT), C, 4 * C);
  k_convT_bf16<<<blocks((long)4 * C * C), 256, 0, stream>>>(mlpW2, Bf(OFF_MLP2WT), 4 * C, C);
  k_convT_bf16<<<blocks(128 * 128), 256, 0, stream>>>(hoW1, Bf(OFF_HO1WT), 128, 128);
  k_convT_bf16<<<blocks(128 * 64), 256, 0, stream>>>(hoW2, Bf(OFF_HO2WT), 128, 64);
  k_conv_bf16<<<blocks((long)LVL * KC * C), 256, 0, stream>>>(vq, Bf(OFF_EBF), (long)LVL * KC * C);
  k_enorm<<<LVL * KC, 256, 0, stream>>>(vq, F(OFF_ENORM));

  // ---- card passing ----
  k_ln1024<<<NT, 256, 0, stream>>>(x, ln1g, ln1b, nullptr, nullptr, F(OFF_BUF1), Bf(OFF_ABF));
  gemm(Bf(OFF_ABF), Bf(OFF_MARKWT), markb, 1.f, F(OFF_BUF2), nullptr, NT, C, C, 0, 0);
  gemm(Bf(OFF_ABF), Bf(OFF_GATEWT), gateb, 1.f, F(OFF_BUF3), nullptr, NT, C, C, 1, 0);
  k_mul<<<blocks((long)NTC), 256, 0, stream>>>(F(OFF_BUF2), F(OFF_BUF3), (long)NTC);
  k_cumsum<<<256, 256, 0, stream>>>(F(OFF_BUF2), F(OFF_CSUM));
  k_carry<<<16, 256, 0, stream>>>(F(OFF_CSUM), F(OFF_CARR));
  k_ncar<<<128, 256, 0, stream>>>(F(OFF_CARR), carg, carb, F(OFF_NCAR));
  k_combined<<<32768, 256, 0, stream>>>(F(OFF_BUF1), F(OFF_BUF2), F(OFF_NCAR), cdg, cdb, Bf(OFF_HIDR));
  gemm(Bf(OFF_HIDR), Bf(OFF_HO1WT), hob1, 1.f, nullptr, Bf(OFF_H1R), NT * H, 128, 128, 2, 0);
  gemm(Bf(OFF_H1R), Bf(OFF_HO2WT), hob2, 1.f, nullptr, Bf(OFF_ATTR), NT * H, 64, 128, 0, 1);
  gemm(Bf(OFF_ATTR), Bf(OFF_PROJWT), projb, 1.f, F(OFF_BUF2), nullptr, NT, C, C, 0, 0);
  // x1 = x + ln1_out + LN(proj_out)
  k_ln1024<<<NT, 256, 0, stream>>>(F(OFF_BUF2), alng, alnb, x, F(OFF_BUF1), F(OFF_BUF3), nullptr);

  // ---- MLP ----
  k_ln1024<<<NT, 256, 0, stream>>>(F(OFF_BUF3), ln2g, ln2b, nullptr, nullptr, nullptr, Bf(OFF_ABF));
  gemm(Bf(OFF_ABF), Bf(OFF_MLP1WT), mlpb1, 1.f, nullptr, Bf(OFF_HIDR), NT, 4 * C, C, 2, 0);
  gemm(Bf(OFF_HIDR), Bf(OFF_MLP2WT), mlpb2, 1.f, F(OFF_BUF2), nullptr, NT, C, 4 * C, 0, 0);
  k_add<<<blocks((long)NTC), 256, 0, stream>>>(F(OFF_BUF3), F(OFF_BUF2), F(OFF_H1R), (long)NTC); // x2

  // ---- RVQ ----
  k_ln1024<<<NT, 256, 0, stream>>>(F(OFF_H1R), ln3g, ln3b, nullptr, nullptr, F(OFF_HIDR), Bf(OFF_ATTR));
  hipMemsetAsync(ws + OFF_BUF2, 0, NTC * sizeof(float), stream);  // total = 0
  hipMemsetAsync(ws + OFF_LOSS, 0, sizeof(float), stream);
  for (int l = 0; l < LVL; ++l) {
    gemm(Bf(OFF_ATTR), Bf(OFF_EBF) + (size_t)l * KC * C, F(OFF_ENORM) + l * KC, -2.f,
         F(OFF_BUF1), nullptr, NT, KC, C, 0, 0);                  // scores
    k_argmin<<<NT / 8, 256, 0, stream>>>(F(OFF_BUF1), (int*)(ws + OFF_IDX), NT, KC);
    k_rvq_update<<<blocks((long)NTC), 256, 0, stream>>>(
        vq + (size_t)l * KC * C, (int*)(ws + OFF_IDX),
        F(OFF_HIDR), Bf(OFF_ATTR), F(OFF_BUF2), F(OFF_LOSS), (long)NTC);
  }

  // ---- output: x2 + total, then both losses ----
  k_add<<<blocks((long)NTC), 256, 0, stream>>>(F(OFF_H1R), F(OFF_BUF2), (float*)d_out, (long)NTC);
  k_finalize_loss<<<1, 32, 0, stream>>>(F(OFF_LOSS), (float*)d_out, (long)NTC);
}
